// MPGG_36979668418588
// MI455X (gfx1250) — matrix operand order repaired and checked by reading, not compiled, not run on hardware
//
#include <hip/hip_runtime.h>
#include <hip/hip_bf16.h>
#include <math.h>

typedef __attribute__((ext_vector_type(16))) _Float16 v16h;
typedef __attribute__((ext_vector_type(8)))  _Float16 v8h;
typedef __attribute__((ext_vector_type(8)))  float    v8f;

#define NNODES 768
#define LPN 64
#define EDIM 64
#define EHALF 294528   /* 768*767/2 */

// ---------------------------------------------------------------------------
// Per-node precompute:
//   P[n][0:128] = node[n] @ eg_w1[0:64,:]          (src half)   [optional]
//   Q[n][0:128] = node[n] @ eg_w1[64:128,:]        (dst half)   [optional]
//   L[n][0:64]  = node[n] @ lin_w + lin_b
//   S[n][0:32]  = node[n] @ att_w1[64:128,:]       (x_src half of att MLP)
// ---------------------------------------------------------------------------
__global__ __launch_bounds__(64) void node_pre(
    const float* __restrict__ X, const float* __restrict__ egw1,
    const float* __restrict__ linw, const float* __restrict__ linb,
    const float* __restrict__ attw1,
    float* __restrict__ P, float* __restrict__ Q,
    float* __restrict__ L, float* __restrict__ S)
{
  const int n = blockIdx.x;
  const int t = threadIdx.x;
  __shared__ float xn[64];
  xn[t] = X[n * 64 + t];
  __syncthreads();

  if (P != nullptr) {
    for (int j = t; j < 128; j += 64) {
      float ap = 0.f, aq = 0.f;
      #pragma unroll 8
      for (int k = 0; k < 64; ++k) {
        ap += xn[k] * egw1[k * 128 + j];
        aq += xn[k] * egw1[(64 + k) * 128 + j];
      }
      P[n * 128 + j] = ap;
      Q[n * 128 + j] = aq;
    }
  }
  {
    float acc = linb[t];
    #pragma unroll 8
    for (int k = 0; k < 64; ++k) acc += xn[k] * linw[k * 64 + t];
    L[n * 64 + t] = acc;
  }
  if (t < 32) {
    float acc = 0.f;
    #pragma unroll 8
    for (int k = 0; k < 64; ++k) acc += xn[k] * attw1[(64 + k) * 32 + t];
    S[n * 32 + t] = acc;
  }
}

// edge index in the reference's [triu ; reversed-triu] ordering
__device__ __forceinline__ int edge_idx(int s, int d) {
  if (s < d) return s * 767 - (s * (s - 1)) / 2 + (d - s - 1);
  return EHALF + d * 767 - (d * (d - 1)) / 2 + (s - d - 1);
}

// A-matrix fragment (16x32 f16): lane = M row + K-half select (ISA 7.12.2)
__device__ __forceinline__ v16h load_fragA(const _Float16* base, int kStride,
                                           int kTile, int lane) {
  const int row = lane & 15, sel = lane >> 4;
  const _Float16* p = base + row * kStride + kTile * 32 + sel * 8;
  v8h lo = *(const v8h*)p;
  v8h hi = *(const v8h*)(p + 16);
  v16h r;
  #pragma unroll
  for (int i = 0; i < 8; ++i) { r[i] = lo[i]; r[i + 8] = hi[i]; }
  return r;
}

// B-matrix fragment (32x16 f16): lane = N col + K-half select (same K order as A)
// base points at transposed weights laid out [N][K] in LDS.
__device__ __forceinline__ v16h load_fragB(const _Float16* base, int kStride,
                                           int kTile, int lane) {
  const int n = lane & 15, sel = lane >> 4;
  const _Float16* p = base + n * kStride + kTile * 32 + sel * 8;
  v8h lo = *(const v8h*)p;
  v8h hi = *(const v8h*)(p + 16);
  v16h r;
  #pragma unroll
  for (int i = 0; i < 8; ++i) { r[i] = lo[i]; r[i + 8] = hi[i]; }
  return r;
}

// ---------------------------------------------------------------------------
// Fused per-edge conv pass. One workgroup per destination node d (no atomics).
// 4 waves; each wave owns a 16-source group end-to-end:
//   hidden = relu(P[s]+Qd)                 -> LDS f16  [16x128]
//   edge   = hidden @ W2t + b2  (16 WMMA, C chains start from inline 0)
//   ah     = relu(edge @ At + S + b)  (4 WMMA, B in regs, inline-0 C)
//   att    = sigmoid(ah @ aw2 + b2)   (VALU)
//   msg   += att * L[s]               (VALU, per-wave accum)
// ---------------------------------------------------------------------------
__global__ __launch_bounds__(128, 1) void conv_pass(
    const float* __restrict__ P, const float* __restrict__ Q,
    const float* __restrict__ Lg, const float* __restrict__ Sg,
    const float* __restrict__ eg_b1, const float* __restrict__ eg_w2,
    const float* __restrict__ eg_b2, const float* __restrict__ att_w1,
    const float* __restrict__ att_b1, const float* __restrict__ att_w2,
    const float* __restrict__ att_b2,
    float* __restrict__ Xout, float* __restrict__ attOut, int doRelu)
{
  const int d    = blockIdx.x;
  const int tid  = threadIdx.x;
  const int lane = tid & 31;
  const int w    = tid >> 5;

  __shared__ __align__(16) _Float16 sW2t[64 * 128];   // eg_w2^T  [N=64][K=128]
  __shared__ __align__(16) _Float16 sAt[32 * 64];     // att_w1[:64]^T [N=32][K=64]
  __shared__ float sQd[128];                          // Q[d] + eg_b1 (folded)
  __shared__ float sB2e[64];
  __shared__ float sAb1[32];
  __shared__ float sAw2[32];
  __shared__ __align__(16) _Float16 sHid[4][16 * 128];
  __shared__ __align__(16) _Float16 sEdge[4][16 * 64];
  __shared__ __align__(16) float sS[4][16 * 32];
  __shared__ __align__(16) float sL[4][16 * 64];
  __shared__ float sAh[4][16 * 32];
  __shared__ float sAtt[4][16];
  __shared__ float sRed[4][64];

  // ---- shared init (transpose weights to f16, fold biases) ----
  for (int idx = tid; idx < 64 * 128; idx += 128) {
    int n = idx >> 7, k = idx & 127;
    sW2t[idx] = (_Float16)eg_w2[k * 64 + n];
  }
  for (int idx = tid; idx < 32 * 64; idx += 128) {
    int n = idx >> 6, k = idx & 63;
    sAt[idx] = (_Float16)att_w1[k * 32 + n];
  }
  if (tid < 128) sQd[tid] = Q[d * 128 + tid] + eg_b1[tid];
  if (tid < 64)  sB2e[tid] = eg_b2[tid];
  if (tid < 32) { sAb1[tid] = att_b1[tid]; sAw2[tid] = att_w2[tid]; }
  __syncthreads();

  // attention-layer B fragments are loop-invariant: keep in VGPRs
  v16h bAtt[2][2];
  #pragma unroll
  for (int nb = 0; nb < 2; ++nb)
    #pragma unroll
    for (int kt = 0; kt < 2; ++kt)
      bAtt[nb][kt] = load_fragB(sAt + nb * 16 * 64, 64, kt, lane);

  const float attb2 = att_b2[0];
  float m0 = 0.f, m1 = 0.f;          // per-lane message accumulators (2 ch)
  const int col = lane & 15, sel = lane >> 4;

  for (int g = w; g < 48; g += 4) {
    const int s0 = g * 16;
    // prefetch next group's P rows (hint only)
    {
      int pn = s0 + 64; if (pn > 752) pn = 752;
      __builtin_prefetch(&P[pn * 128 + (lane << 4)], 0, 1);
    }
    // stage L[16x64], S[16x32] for this source group (contiguous, float4)
    for (int idx = lane; idx < 256; idx += 32)
      *(float4*)&sL[w][idx * 4] = *(const float4*)&Lg[s0 * 64 + idx * 4];
    for (int idx = lane; idx < 128; idx += 32)
      *(float4*)&sS[w][idx * 4] = *(const float4*)&Sg[s0 * 32 + idx * 4];

    // hidden = relu(P[s] + Q[d] + b1): float2 loads, packed 2xf16 stores
    for (int idx = lane; idx < 1024; idx += 32) {
      int r = idx >> 6, c = (idx & 63) * 2;
      float2 pv = *(const float2*)&P[(s0 + r) * 128 + c];
      float2 qv = *(const float2*)&sQd[c];
      union { _Float16 h[2]; unsigned u; } pk;
      pk.h[0] = (_Float16)fmaxf(pv.x + qv.x, 0.f);
      pk.h[1] = (_Float16)fmaxf(pv.y + qv.y, 0.f);
      *(unsigned*)&sHid[w][r * 128 + c] = pk.u;
    }

    // ---- edge features: [16x128] @ [128x64], kt outer / nb inner ----
    // Accumulator chains start from zero (inline SRC2=0); bias folded into
    // the f16 convert at store time.
    v8f accE[4] = {};
    #pragma unroll
    for (int kt = 0; kt < 4; ++kt) {
      v16h a = load_fragA(sHid[w], 128, kt, lane);   // one A per K-tile
      #pragma unroll
      for (int nb = 0; nb < 4; ++nb) {
        v16h b = load_fragB(sW2t + nb * 16 * 128, 128, kt, lane);
        accE[nb] = __builtin_amdgcn_wmma_f32_16x16x32_f16(
            false, a, false, b, (short)0, accE[nb], false, false);
      }
    }
    #pragma unroll
    for (int nb = 0; nb < 4; ++nb) {
      const float cinit = sB2e[nb * 16 + col];
      #pragma unroll
      for (int j = 0; j < 8; ++j)
        sEdge[w][(j + sel * 8) * 64 + nb * 16 + col] =
            (_Float16)(accE[nb][j] + cinit);
    }

    // ---- attention hidden: [16x64] @ [64x32], B fragments in registers ----
    v8f accA[2] = {};
    #pragma unroll
    for (int kt = 0; kt < 2; ++kt) {
      v16h a = load_fragA(sEdge[w], 64, kt, lane);
      #pragma unroll
      for (int nb = 0; nb < 2; ++nb)
        accA[nb] = __builtin_amdgcn_wmma_f32_16x16x32_f16(
            false, a, false, bAtt[nb][kt], (short)0, accA[nb], false, false);
    }
    #pragma unroll
    for (int nb = 0; nb < 2; ++nb) {
      const float b1 = sAb1[nb * 16 + col];
      #pragma unroll
      for (int j = 0; j < 8; ++j) {
        float v = accA[nb][j] + sS[w][(j + sel * 8) * 32 + nb * 16 + col] + b1;
        sAh[w][(j + sel * 8) * 32 + nb * 16 + col] = fmaxf(v, 0.f);
      }
    }

    // ---- logit (32 -> 1), sigmoid, write attention ----
    if (lane < 16) {
      const int s = s0 + lane;
      float x = attb2;
      #pragma unroll
      for (int k = 0; k < 32; ++k) x += sAh[w][lane * 32 + k] * sAw2[k];
      float a = (s == d) ? 0.f : (1.f / (1.f + __expf(-x)));
      sAtt[w][lane] = a;
      if (attOut != nullptr && s != d) attOut[edge_idx(s, d)] = a;
    }

    // ---- message accumulation: msg += att * L[s] ----
    const int c0 = lane * 2;
    #pragma unroll
    for (int r = 0; r < 16; ++r) {
      float a = sAtt[w][r];
      float2 lv = *(const float2*)&sL[w][r * 64 + c0];
      m0 += a * lv.x;
      m1 += a * lv.y;
    }
  }

  // ---- cross-wave reduction of the segment sum for node d ----
  sRed[w][lane * 2]     = m0;
  sRed[w][lane * 2 + 1] = m1;
  __syncthreads();
  if (tid < 64) {
    float x = sRed[0][tid] + sRed[1][tid] + sRed[2][tid] + sRed[3][tid];
    if (doRelu) x = fmaxf(x, 0.f);
    Xout[d * 64 + tid] = x;
  }
}

extern "C" void kernel_launch(void* const* d_in, const int* in_sizes, int n_in,
                              void* d_out, int out_size, void* d_ws, size_t ws_size,
                              hipStream_t stream) {
  const float* z       = (const float*)d_in[0];
  const float* eg_w1   = (const float*)d_in[1];
  const float* eg_b1   = (const float*)d_in[2];
  const float* eg_w2   = (const float*)d_in[3];
  const float* eg_b2   = (const float*)d_in[4];
  const float* att0_w1 = (const float*)d_in[5];
  const float* att0_b1 = (const float*)d_in[6];
  const float* att0_w2 = (const float*)d_in[7];
  const float* att0_b2 = (const float*)d_in[8];
  const float* lin0_w  = (const float*)d_in[9];
  const float* lin0_b  = (const float*)d_in[10];
  const float* att1_w1 = (const float*)d_in[11];
  const float* att1_b1 = (const float*)d_in[12];
  const float* att1_w2 = (const float*)d_in[13];
  const float* att1_b2 = (const float*)d_in[14];
  const float* lin1_w  = (const float*)d_in[15];
  const float* lin1_b  = (const float*)d_in[16];

  float* ws = (float*)d_ws;
  float* P  = ws;                     // 768*128
  float* Q  = P  + NNODES * 128;      // 768*128
  float* L  = Q  + NNODES * 128;      // 768*64 (reused by both passes)
  float* S  = L  + NNODES * 64;       // 768*32 (reused)
  float* X0 = S  + NNODES * 32;       // 768*64

  float* outX   = (float*)d_out;            // [768,64]
  float* outAtt = outX + NNODES * LPN;      // [2E]

  // Pass 0
  node_pre<<<NNODES, 64, 0, stream>>>(z, eg_w1, lin0_w, lin0_b, att0_w1,
                                      P, Q, L, S);
  conv_pass<<<NNODES, 128, 0, stream>>>(P, Q, L, S, eg_b1, eg_w2, eg_b2,
                                        att0_w1, att0_b1, att0_w2, att0_b2,
                                        X0, nullptr, /*doRelu=*/1);
  // Pass 1 (edges recomputed from shared P/Q; only L/S depend on x)
  node_pre<<<NNODES, 64, 0, stream>>>(X0, nullptr, lin1_w, lin1_b, att1_w1,
                                      nullptr, nullptr, L, S);
  conv_pass<<<NNODES, 128, 0, stream>>>(P, Q, L, S, eg_b1, eg_w2, eg_b2,
                                        att1_w1, att1_b1, att1_w2, att1_b2,
                                        outX, outAtt, /*doRelu=*/0);
}
